// SelfAttentionHead_48395691492066
// MI455X (gfx1250) — matrix operand order repaired and checked by reading, not compiled, not run on hardware
//
#include <hip/hip_runtime.h>

#ifndef __has_builtin
#define __has_builtin(x) 0
#endif

#if defined(__HIP_DEVICE_COMPILE__) && __has_builtin(__builtin_amdgcn_tensor_load_to_lds)
#define HAVE_TDM 1
#else
#define HAVE_TDM 0
#endif

// ---------- types ----------
typedef __attribute__((ext_vector_type(16))) __bf16 v16bf;
typedef __attribute__((ext_vector_type(8)))  __bf16 v8bf;
typedef __attribute__((ext_vector_type(8)))  float  v8f;
typedef __attribute__((ext_vector_type(4)))  unsigned uint32x4;
typedef __attribute__((ext_vector_type(8)))  int      int32x8;
typedef __attribute__((ext_vector_type(4)))  int      int32x4;

union ABfrag { v16bf v; v8bf h[2]; };

#define NEMB 1024
#define HS   64
#define TSEQ 2048
#define NB   8
#define BTROWS (NB * TSEQ)   // 16384

__device__ __forceinline__ __bf16 f2bf(float f) {
  unsigned u = __builtin_bit_cast(unsigned, f);
  unsigned r = u + 0x7FFFu + ((u >> 16) & 1u);   // round-to-nearest-even
  unsigned short hs = (unsigned short)(r >> 16);
  return __builtin_bit_cast(__bf16, hs);
}

// pack two fp32 -> packed bf16x2 dword (RNE)
__device__ __forceinline__ unsigned pack2bf(float lo, float hi) {
  unsigned ul = __builtin_bit_cast(unsigned, lo);
  unsigned uh = __builtin_bit_cast(unsigned, hi);
  ul = ul + 0x7FFFu + ((ul >> 16) & 1u);
  uh = uh + 0x7FFFu + ((uh >> 16) & 1u);
  return (ul >> 16) | (uh & 0xFFFF0000u);
}

__device__ __forceinline__ v8f zero8() {
  v8f z;
#pragma unroll
  for (int i = 0; i < 8; ++i) z[i] = 0.0f;
  return z;
}

__device__ __forceinline__ v8f wmma_bf16(v16bf a, v16bf b, v8f c) {
  // D = A(16x32 bf16) * B(32x16 bf16) + C(16x16 f32)
  return __builtin_amdgcn_wmma_f32_16x16x32_bf16(false, a, false, b,
                                                 (short)0, c, false, false);
}

#if HAVE_TDM
// low 32 bits of a generic pointer to a __shared__ object = LDS byte offset
__device__ __forceinline__ unsigned lds_off(const void* p) {
  return (unsigned)(unsigned long long)(uintptr_t)p;
}

__device__ __forceinline__ void tdm_wait0() {
#if __has_builtin(__builtin_amdgcn_s_wait_tensorcnt)
  __builtin_amdgcn_s_wait_tensorcnt(0);
#else
  asm volatile("s_wait_tensorcnt 0x0" ::: "memory");
#endif
}

// TDM: load a [rows x 64] bf16 tile (row-major, tensor row stride 64) into LDS
// with pitch 72 bf16 (pad 4 dwords after every 32 dwords => 144B rows).
__device__ __forceinline__ void tdm_load_tile64(unsigned lds_byte,
                                                const unsigned short* gsrc,
                                                unsigned rows,
                                                unsigned tensor_rows) {
  unsigned long long ga = (unsigned long long)(uintptr_t)gsrc;
  uint32x4 g0;
  g0[0] = 1u;                                      // count=1, user mode
  g0[1] = lds_byte;                                // lds_addr
  g0[2] = (unsigned)(ga & 0xFFFFFFFFu);            // global_addr[31:0]
  g0[3] = (unsigned)((ga >> 32) & 0x01FFFFFFu)     // global_addr[56:32]
        | (2u << 30);                              // type = 2 ("image")
  int32x8 g1;
  g1[0] = (int)((1u << 16)      // data_size = 2 bytes
              | (1u << 20)      // pad_enable
              | (4u << 22)      // pad_interval: 32 dwords (one 128B row)
              | (3u << 25));    // pad_amount:   4 dwords (16B)
  g1[1] = (int)(64u << 16);                 // tensor_dim0[15:0] = 64
  g1[2] = (int)((tensor_rows & 0xFFFFu) << 16);  // dim0 hi=0 | tensor_dim1 lo16
  g1[3] = (int)((tensor_rows >> 16) | (64u << 16)); // tensor_dim1 hi | tile_dim0=64
  g1[4] = (int)rows;                        // tile_dim1 = rows, tile_dim2 = 0
  g1[5] = 64;                               // tensor_dim0_stride[31:0] = 64
  g1[6] = 0;                                // stride hi | tensor_dim1_stride lo
  g1[7] = 0;
  int32x4 z4 = {0, 0, 0, 0};
#if __clang_major__ >= 23
  int32x8 z8 = {0, 0, 0, 0, 0, 0, 0, 0};
  __builtin_amdgcn_tensor_load_to_lds(g0, g1, z4, z4, z8, 0);
#else
  __builtin_amdgcn_tensor_load_to_lds(g0, g1, z4, z4, 0);
#endif
}
#endif  // HAVE_TDM

// =====================================================================
// Kernel 1: projections  K/Q/V = x @ W + b   (output stored as bf16)
//   grid.x = 128 blocks, 256 threads (8 waves)
//   K-chunk = 64 per iteration: 2 A-frags, 24 WMMAs per iteration
// =====================================================================
__global__ __launch_bounds__(256)
void proj_kernel(const float* __restrict__ x,
                 const float* __restrict__ Wk, const float* __restrict__ bk,
                 const float* __restrict__ Wq, const float* __restrict__ bq,
                 const float* __restrict__ Wv, const float* __restrict__ bv,
                 unsigned short* __restrict__ Kw,
                 unsigned short* __restrict__ Qw,
                 unsigned short* __restrict__ Vw)
{
  __shared__ __bf16 At[128 * 72];       // x tile, row-major, pitch 72 (16B-aligned rows)
  __shared__ __bf16 Wt[3][64 * 64];     // W chunk, column-major [h][k], pitch 64

  const int tid  = threadIdx.x;
  const int lane = tid & 31;
  const int wave = tid >> 5;
  const int row0 = blockIdx.x * 128;

  v8f accK[4], accQ[4], accV[4];
#pragma unroll
  for (int t = 0; t < 4; ++t) { accK[t] = zero8(); accQ[t] = zero8(); accV[t] = zero8(); }

  const int mrow  = wave * 16 + (lane & 15);
  const int khalf = (lane < 16) ? 0 : 8;    // unused (A-fragments use kb16)
  const int kb16  = (lane < 16) ? 0 : 16;   // B-fragment K sub-base

  for (int c0 = 0; c0 < NEMB; c0 += 64) {
    __syncthreads();
    // ---- stage x tile: float2 loads, packed bf16x2 dword stores ----
    for (int e = tid; e < 128 * 32; e += 256) {
      int r = e >> 5, cc = (e & 31) * 2;
      float2 xv = *(const float2*)&x[(size_t)(row0 + r) * NEMB + c0 + cc];
      *(unsigned*)&At[r * 72 + cc] = pack2bf(xv.x, xv.y);
    }
    // ---- stage W chunks column-major, packed pairs along k ----
    for (int e = tid; e < 64 * 32; e += 256) {
      int h = e & 63, kk = (e >> 6) * 2;
      const size_t o0 = (size_t)(c0 + kk) * HS + h;
      *(unsigned*)&Wt[0][h * 64 + kk] = pack2bf(Wk[o0], Wk[o0 + HS]);
      *(unsigned*)&Wt[1][h * 64 + kk] = pack2bf(Wq[o0], Wq[o0 + HS]);
      *(unsigned*)&Wt[2][h * 64 + kk] = pack2bf(Wv[o0], Wv[o0 + HS]);
    }
    // ---- prefetch next x chunk (one 128B line per thread) ----
    if (c0 + 64 < NEMB) {
      const float* nx = &x[(size_t)(row0 + (tid & 127)) * NEMB + c0 + 64 + (tid >> 7) * 32];
      __builtin_prefetch(nx, 0, 1);
    }
    __syncthreads();

    // ---- A fragments for both K=32 sub-chunks ----
    ABfrag a[2];
#pragma unroll
    for (int c = 0; c < 2; ++c) {
      const __bf16* arow = &At[mrow * 72 + c * 32];
      a[c].h[0] = *(const v8bf*)(arow + kb16);
      a[c].h[1] = *(const v8bf*)(arow + kb16 + 8);
    }

#pragma unroll
    for (int c = 0; c < 2; ++c) {
      // preload all 12 B fragments for this sub-chunk (clause the ds loads)
      ABfrag Bf[12];
#pragma unroll
      for (int nt = 0; nt < 4; ++nt) {
        const int n = nt * 16 + (lane & 15);
#pragma unroll
        for (int m = 0; m < 3; ++m) {
          const __bf16* bp = &Wt[m][n * 64 + c * 32 + kb16];
          Bf[nt * 3 + m].h[0] = *(const v8bf*)(bp);
          Bf[nt * 3 + m].h[1] = *(const v8bf*)(bp + 8);
        }
      }
      // back-to-back WMMA chain
#pragma unroll
      for (int nt = 0; nt < 4; ++nt) {
        accK[nt] = wmma_bf16(a[c].v, Bf[nt * 3 + 0].v, accK[nt]);
        accQ[nt] = wmma_bf16(a[c].v, Bf[nt * 3 + 1].v, accQ[nt]);
        accV[nt] = wmma_bf16(a[c].v, Bf[nt * 3 + 2].v, accV[nt]);
      }
    }
  }

  // ---- epilogue: bias add, convert to bf16, store to workspace ----
  const int rb = wave * 16 + ((lane < 16) ? 0 : 8);
#pragma unroll
  for (int nt = 0; nt < 4; ++nt) {
    const int n = nt * 16 + (lane & 15);
    const float bkv = bk[n], bqv = bq[n], bvv = bv[n];
#pragma unroll
    for (int r = 0; r < 8; ++r) {
      const size_t off = (size_t)(row0 + rb + r) * HS + n;
      Kw[off] = __builtin_bit_cast(unsigned short, f2bf(accK[nt][r] + bkv));
      Qw[off] = __builtin_bit_cast(unsigned short, f2bf(accQ[nt][r] + bqv));
      Vw[off] = __builtin_bit_cast(unsigned short, f2bf(accV[nt][r] + bvv));
    }
  }
}

// =====================================================================
// Kernel 2: fused causal softplus-attention
//   out[b,t,:] = sum_{s<=t} softplus(scale * k[t].q[s]) * v[b,s,:]
//   grid = (16 t-tiles, 8 batches), 256 threads (8 waves)
//   K/Q tiles staged by the Tensor Data Mover when available.
// =====================================================================
__global__ __launch_bounds__(256)
void attn_kernel(const unsigned short* __restrict__ Kw,
                 const unsigned short* __restrict__ Qw,
                 const unsigned short* __restrict__ Vw,
                 float* __restrict__ out)
{
  __shared__ __bf16 Kst[128 * 72];        // K tile, row-major pitch 72
  __shared__ __bf16 Qst[128 * 72];        // Q tile, row-major pitch 72
  __shared__ __bf16 Vt [64 * 136];        // V tile transposed [h][s], pitch 136
  __shared__ __bf16 Pst[8 * 16 * 128];    // per-wave P tile (16 x 128)

  const int tid  = threadIdx.x;
  const int lane = tid & 31;
  const int wave = tid >> 5;
  const int i    = blockIdx.x;            // t-tile index
  const int b    = blockIdx.y;            // batch

  const unsigned short* Kb = Kw + (size_t)b * TSEQ * HS;
  const unsigned short* Qb = Qw + (size_t)b * TSEQ * HS;
  const unsigned short* Vb = Vw + (size_t)b * TSEQ * HS;

  // ---- stage K tile for this t-block ----
#if HAVE_TDM
  if (wave == 0) {
    tdm_load_tile64(lds_off(Kst), Kb + (size_t)(128 * i) * HS, 128, TSEQ);
    tdm_wait0();
  }
#else
  for (int e = tid; e < 128 * 16; e += 256) {
    int r = e >> 4, p = e & 15;
    *(uint2*)&Kst[r * 72 + 4 * p] =
        *(const uint2*)&Kb[(size_t)(128 * i + r) * HS + 4 * p];
  }
#endif
  __syncthreads();

  // ---- K A-fragments (fixed for whole block) ----
  const int mrow  = wave * 16 + (lane & 15);
  const int khalf = (lane < 16) ? 0 : 8;
  const int kb16  = (lane < 16) ? 0 : 16;
  ABfrag ak[2];
#pragma unroll
  for (int c = 0; c < 2; ++c) {
    const __bf16* arow = &Kst[mrow * 72 + c * 32];
    ak[c].h[0] = *(const v8bf*)(arow + kb16);
    ak[c].h[1] = *(const v8bf*)(arow + kb16 + 8);
  }

  v8f accO[4];
#pragma unroll
  for (int t = 0; t < 4; ++t) accO[t] = zero8();

  const float scale = 0.03125f;           // 1024^-0.5
  const int   rb    = (lane < 16) ? 0 : 8;
  __bf16* Pw = &Pst[wave * 16 * 128];

  for (int j = 0; j <= i; ++j) {
    __syncthreads();   // protect Qst/Vt from previous iteration's readers

    // ---- stage Q tile: TDM copy (pitch-72 via D# padding) if available ----
#if HAVE_TDM
    if (wave == 0) {
      tdm_load_tile64(lds_off(Qst), Qb + (size_t)(128 * j) * HS, 128, TSEQ);
    }
    // V tile staged manually (TDM cannot transpose)
    for (int e = tid; e < 128 * 16; e += 256) {
      int r = e >> 4, p = e & 15;
      uint2 vv = *(const uint2*)&Vb[(size_t)(128 * j + r) * HS + 4 * p];
      Vt[(4 * p + 0) * 136 + r] = __builtin_bit_cast(__bf16, (unsigned short)(vv.x & 0xFFFFu));
      Vt[(4 * p + 1) * 136 + r] = __builtin_bit_cast(__bf16, (unsigned short)(vv.x >> 16));
      Vt[(4 * p + 2) * 136 + r] = __builtin_bit_cast(__bf16, (unsigned short)(vv.y & 0xFFFFu));
      Vt[(4 * p + 3) * 136 + r] = __builtin_bit_cast(__bf16, (unsigned short)(vv.y >> 16));
    }
#else
    for (int e = tid; e < 128 * 16; e += 256) {
      int r = e >> 4, p = e & 15;
      *(uint2*)&Qst[r * 72 + 4 * p] =
          *(const uint2*)&Qb[(size_t)(128 * j + r) * HS + 4 * p];
      uint2 vv = *(const uint2*)&Vb[(size_t)(128 * j + r) * HS + 4 * p];
      Vt[(4 * p + 0) * 136 + r] = __builtin_bit_cast(__bf16, (unsigned short)(vv.x & 0xFFFFu));
      Vt[(4 * p + 1) * 136 + r] = __builtin_bit_cast(__bf16, (unsigned short)(vv.x >> 16));
      Vt[(4 * p + 2) * 136 + r] = __builtin_bit_cast(__bf16, (unsigned short)(vv.y & 0xFFFFu));
      Vt[(4 * p + 3) * 136 + r] = __builtin_bit_cast(__bf16, (unsigned short)(vv.y >> 16));
    }
#endif
    // ---- prefetch next tile's Q/V lines while this one computes ----
    if (j < i) {
      const unsigned short* qn = &Qb[(size_t)(128 * (j + 1) + (tid & 127)) * HS];
      const unsigned short* vn = &Vb[(size_t)(128 * (j + 1) + (tid & 127)) * HS];
      __builtin_prefetch(qn, 0, 1);
      __builtin_prefetch(vn, 0, 1);
    }
#if HAVE_TDM
    if (wave == 0) tdm_wait0();   // Q tile landed before the barrier publishes it
#endif
    __syncthreads();

    // ---- S = K_tile @ Q_tile^T : preload 8 B-frags, then 8 WMMAs ----
    v8f s[8];
#pragma unroll
    for (int h2 = 0; h2 < 2; ++h2) {
      ABfrag bq[8];
#pragma unroll
      for (int u = 0; u < 4; ++u) {
        const int ct = h2 * 4 + u;
#pragma unroll
        for (int c = 0; c < 2; ++c) {
          const __bf16* bcol = &Qst[(ct * 16 + (lane & 15)) * 72 + c * 32 + kb16];
          bq[u * 2 + c].h[0] = *(const v8bf*)(bcol);
          bq[u * 2 + c].h[1] = *(const v8bf*)(bcol + 8);
        }
      }
#pragma unroll
      for (int u = 0; u < 4; ++u) {
        const int ct = h2 * 4 + u;
        v8f sv = zero8();
        sv = wmma_bf16(ak[0].v, bq[u * 2 + 0].v, sv);
        sv = wmma_bf16(ak[1].v, bq[u * 2 + 1].v, sv);
        s[ct] = sv;
      }
    }

    // ---- mask + softplus, write P into per-wave LDS (A-layout source) ----
#pragma unroll
    for (int ct = 0; ct < 8; ++ct) {
      const int sg = 128 * j + ct * 16 + (lane & 15);
#pragma unroll
      for (int r = 0; r < 8; ++r) {
        const int tg = 128 * i + wave * 16 + rb + r;
        float xv = s[ct][r] * scale;
        float p;
        if (j == i && sg > tg) {
          p = 0.0f;                                 // softplus(-inf) = 0
        } else {
          p = fmaxf(xv, 0.0f) + log1pf(__expf(-fabsf(xv)));
        }
        Pw[(rb + r) * 128 + ct * 16 + (lane & 15)] = f2bf(p);
      }
    }
    // (Pw region is wave-private; LDS ops from the same wave stay ordered)

    // ---- accO += P(16x128) @ V(128x64): preload ap+4 bv per s-chunk ----
#pragma unroll
    for (int c = 0; c < 4; ++c) {
      ABfrag ap, bv[4];
      const __bf16* prow = &Pw[(lane & 15) * 128 + c * 32];
      ap.h[0] = *(const v8bf*)(prow + kb16);
      ap.h[1] = *(const v8bf*)(prow + kb16 + 8);
#pragma unroll
      for (int nt = 0; nt < 4; ++nt) {
        const __bf16* bcol = &Vt[(nt * 16 + (lane & 15)) * 136 + c * 32 + kb16];
        bv[nt].h[0] = *(const v8bf*)(bcol);
        bv[nt].h[1] = *(const v8bf*)(bcol + 8);
      }
#pragma unroll
      for (int nt = 0; nt < 4; ++nt) {
        accO[nt] = wmma_bf16(ap.v, bv[nt].v, accO[nt]);
      }
    }
  }

  // ---- store output (fp32) ----
#pragma unroll
  for (int nt = 0; nt < 4; ++nt) {
    const int n = nt * 16 + (lane & 15);
#pragma unroll
    for (int r = 0; r < 8; ++r) {
      const int tg = 128 * i + wave * 16 + rb + r;
      out[((size_t)b * TSEQ + tg) * HS + n] = accO[nt][r];
    }
  }
}

// =====================================================================
extern "C" void kernel_launch(void* const* d_in, const int* in_sizes, int n_in,
                              void* d_out, int out_size, void* d_ws, size_t ws_size,
                              hipStream_t stream) {
  const float* x  = (const float*)d_in[0];
  const float* Wk = (const float*)d_in[1];
  const float* bk = (const float*)d_in[2];
  const float* Wq = (const float*)d_in[3];
  const float* bq = (const float*)d_in[4];
  const float* Wv = (const float*)d_in[5];
  const float* bv = (const float*)d_in[6];
  float* out = (float*)d_out;

  unsigned short* Kw = (unsigned short*)d_ws;                  // [16384][64] bf16
  unsigned short* Qw = Kw + (size_t)BTROWS * HS;
  unsigned short* Vw = Qw + (size_t)BTROWS * HS;

  proj_kernel<<<BTROWS / 128, 256, 0, stream>>>(x, Wk, bk, Wq, bq, Wv, bv, Kw, Qw, Vw);
  attn_kernel<<<dim3(TSEQ / 128, NB), 256, 0, stream>>>(Kw, Qw, Vw, out);
}
